// MSGIFSR_13632226198285
// MI455X (gfx1250) — matrix operand order repaired and checked by reading, not compiled, not run on hardware
//
#include <hip/hip_runtime.h>
#include <math.h>

// ---------------- sizes ----------------
#define V_  50000
#define D_  256
#define B_  512
#define N_  10240
#define O_  3

// ---------------- workspace layout (bytes) ----------------
#define WS_TARGET   0u                          // bf16 [V,256]      25,600,000
#define WS_FEATSF   25600000u                   // f32  [N,256]      10,485,760
#define WS_FEATSB   36085760u                   // bf16 [N,256]       5,242,880
#define WS_WUT      41328640u                   // bf16 [O,256,256]     393,216
#define WS_FVS      41721856u                   // f32  [O,B,256]     1,572,864
#define WS_FV       43294720u                   // f32  [O,B,256]     1,572,864
#define WS_E        44867584u                   // f32  [O,N]           122,880
#define WS_SRG      44990464u                   // f32  [O,B,256]     1,572,864
#define WS_SRB      46563328u                   // bf16 [O*B,256]       786,432

// ---------------- WMMA types ----------------
typedef __attribute__((ext_vector_type(16))) __bf16       v16bf;
typedef __attribute__((ext_vector_type(8)))  float        v8f;
typedef __attribute__((ext_vector_type(4)))  unsigned int u32x4;

union Frag { v16bf v; u32x4 q[2]; };

#if defined(__has_builtin)
#if __has_builtin(__builtin_amdgcn_sched_group_barrier)
#define SGB(mask, size) __builtin_amdgcn_sched_group_barrier((mask), (size), 0)
#endif
#endif
#ifndef SGB
#define SGB(mask, size)
#endif

__device__ __forceinline__ unsigned short f2bf(float f) {
  unsigned int u = __float_as_uint(f);
  u += 0x7fffu + ((u >> 16) & 1u);          // round-to-nearest-even
  return (unsigned short)(u >> 16);
}

// A fragment (16-bit 16x32): lane<16 holds K {k0..k0+7, k0+16..k0+23},
// lane>=16 holds K {k0+8..k0+15, k0+24..k0+31}; M = lane&15.
__device__ __forceinline__ void loadA(const unsigned short* __restrict__ p,
                                      int ld, int row, int k0, Frag& f) {
  int lane = threadIdx.x & 31;
  int r  = row + (lane & 15);
  int kk = k0 + ((lane >> 4) << 3);
  const u32x4* q = (const u32x4*)(p + (size_t)r * ld + kk);
  f.q[0] = q[0];     // K kk..kk+7
  f.q[1] = q[2];     // K kk+16..kk+23
}

// B fragment (16-bit 32x16, dense): N = lane&15; lane<16 holds K {k0..k0+7, k0+16..k0+23},
// lane>=16 holds K {k0+8..k0+15, k0+24..k0+31}.
__device__ __forceinline__ void loadB(const unsigned short* __restrict__ p,
                                      int ld, int nrow, int k0, int nclamp, Frag& f) {
  int lane = threadIdx.x & 31;
  int r  = nrow + (lane & 15);
  r = (r < nclamp) ? r : (nclamp - 1);
  int kk = k0 + ((lane >> 4) << 3);
  const u32x4* q = (const u32x4*)(p + (size_t)r * ld + kk);
  f.q[0] = q[0];
  f.q[1] = q[2];
}

#define WMMA_BF16(A, Bm, C) \
  __builtin_amdgcn_wmma_f32_16x16x32_bf16(false, (A).v, false, (Bm).v, (short)0, (C), false, false)

// ---------------- 1: normalize full table -> bf16 ----------------
__global__ void k_target(const float* __restrict__ emb, unsigned short* __restrict__ tb) {
  int row = blockIdx.x, t = threadIdx.x;
  __shared__ float red[256];
  float x = emb[(size_t)row * D_ + t];
  red[t] = x * x; __syncthreads();
  for (int s = 128; s > 0; s >>= 1) { if (t < s) red[t] += red[t + s]; __syncthreads(); }
  float scale = 1.0f / fmaxf(sqrtf(red[0]), 1e-12f);
  tb[(size_t)row * D_ + t] = f2bf(x * scale);
}

// ---------------- 2: feats = normalize(emb[iid]) -> f32 + bf16 ----------------
__global__ void k_feats(const float* __restrict__ emb, const int* __restrict__ iid,
                        float* __restrict__ ff, unsigned short* __restrict__ fb) {
  int row = blockIdx.x, t = threadIdx.x;
  __shared__ float red[256];
  int src = iid[row];
  float x = emb[(size_t)src * D_ + t];
  red[t] = x * x; __syncthreads();
  for (int s = 128; s > 0; s >>= 1) { if (t < s) red[t] += red[t + s]; __syncthreads(); }
  float y = x / fmaxf(sqrtf(red[0]), 1e-12f);
  ff[(size_t)row * D_ + t] = y;
  fb[(size_t)row * D_ + t] = f2bf(y);
}

// ---------------- 3: WuT[o][h][d] = bf16(Wu[o][d][h]) ----------------
__global__ void k_wt(const float* __restrict__ Wu, unsigned short* __restrict__ wut) {
  int oh = blockIdx.x, d = threadIdx.x;
  int o = oh >> 8, h = oh & 255;
  wut[((size_t)oh << 8) + d] = f2bf(Wu[((size_t)o << 16) + ((size_t)d << 8) + h]);
}

// ---------------- 4: gather last nodes + fv = feat_vs @ Wv[o] ----------------
__global__ void k_featv(const float* __restrict__ feats, const float* __restrict__ Wv,
                        const int* __restrict__ last, float* __restrict__ fvs,
                        float* __restrict__ fv) {
  int b = blockIdx.x, o = blockIdx.y, t = threadIdx.x;
  __shared__ float row[256];
  int n = last[o * B_ + b];
  float x = feats[(size_t)n * D_ + t];
  row[t] = x;
  fvs[((size_t)(o * B_ + b)) * D_ + t] = x;
  __syncthreads();
  const float* w = Wv + (size_t)o * D_ * D_;
  float acc = 0.f;
  for (int d = 0; d < D_; ++d) acc += row[d] * w[(size_t)d * D_ + t];
  fv[((size_t)(o * B_ + b)) * D_ + t] = acc;
}

// ---------------- 5: zero ----------------
__global__ void k_zero(float* __restrict__ p, int n) {
  int i = blockIdx.x * 256 + threadIdx.x;
  if (i < n) p[i] = 0.f;
}

// ---------------- 6: e[o,n] = sum_h sigmoid(feats@WuT + bu + fv[idx]) * we ----------------
__global__ void __launch_bounds__(128) k_e(
    const unsigned short* __restrict__ featsb, const unsigned short* __restrict__ wut,
    const float* __restrict__ bu, const float* __restrict__ we,
    const float* __restrict__ fv, const int* __restrict__ idx,
    float* __restrict__ e) {
  int o    = blockIdx.y;
  int n0   = blockIdx.x << 4;        // 16 rows per block
  int wave = threadIdx.x >> 5;       // 4 waves, 64 h each
  int lane = threadIdx.x & 31;
  int h0   = wave << 6;
  const unsigned short* bmat = wut + (size_t)o * D_ * D_;

  v8f c[4];
  v8f z = {0.f,0.f,0.f,0.f,0.f,0.f,0.f,0.f};
  #pragma unroll
  for (int j = 0; j < 4; ++j) c[j] = z;

  // software-pipelined, double-buffered fragment loads
  Frag a[2], bfr[2][4];
  loadA(featsb, D_, n0, 0, a[0]);
  #pragma unroll
  for (int j = 0; j < 4; ++j) loadB(bmat, D_, h0 + j * 16, 0, D_, bfr[0][j]);

  #pragma unroll
  for (int kk = 0; kk < 8; ++kk) {
    int cur = kk & 1, nxt = cur ^ 1;
    if (kk < 7) {
      int k = (kk + 1) << 5;
      loadA(featsb, D_, n0, k, a[nxt]);
      #pragma unroll
      for (int j = 0; j < 4; ++j) loadB(bmat, D_, h0 + j * 16, k, D_, bfr[nxt][j]);
    }
    #pragma unroll
    for (int j = 0; j < 4; ++j) c[j] = WMMA_BF16(a[cur], bfr[cur][j], c[j]);
  }

  // Pin the pipeline: keep >=1 K-step of loads (10 b128) in flight ahead of
  // each WMMA pack. 80 VMEM reads total, 32 WMMAs total.
  SGB(0x020, 20);                     // loads for k0 + k1
  #pragma unroll
  for (int kk = 0; kk < 8; ++kk) {
    SGB(0x008, 4);                    // WMMAs of step kk
    if (kk < 6) SGB(0x020, 10);       // loads for step kk+2
  }

  int nsub = lane & 15;
  int hi8  = (lane >> 4) << 3;
  float buv[4], wev[4];
  #pragma unroll
  for (int j = 0; j < 4; ++j) {
    int h = h0 + j * 16 + nsub;
    buv[j] = bu[o * D_ + h];
    wev[j] = we[o * D_ + h];
  }
  #pragma unroll
  for (int r = 0; r < 8; ++r) {
    int m   = n0 + r + hi8;
    int seg = idx[m];
    const float* fvrow = fv + ((size_t)(o * B_ + seg)) * D_ + h0 + nsub;
    float s = 0.f;
    #pragma unroll
    for (int j = 0; j < 4; ++j) {
      float zv = c[j][r] + buv[j] + fvrow[j * 16];
      float sg = 1.0f / (1.0f + __expf(-zv));
      s += sg * wev[j];
    }
    // reduce over the 16 lanes of this half-wave (same row set)
    s += __shfl_xor(s, 1, 16);
    s += __shfl_xor(s, 2, 16);
    s += __shfl_xor(s, 4, 16);
    s += __shfl_xor(s, 8, 16);
    if (nsub == 0) atomicAdd(&e[o * N_ + m], s);
  }
}

// ---------------- 7: segment softmax + weighted segment sum ----------------
__global__ void k_srg(const float* __restrict__ e, const int* __restrict__ idx,
                      const float* __restrict__ feats, float* __restrict__ srg) {
  int b = blockIdx.x, o = blockIdx.y, t = threadIdx.x;
  __shared__ float red[256];
  __shared__ int sb0, sb1;
  __shared__ float mv, sv;
  if (t == 0) {
    int lo = 0, hi = N_;
    while (lo < hi) { int mid = (lo + hi) >> 1; if (idx[mid] < b) lo = mid + 1; else hi = mid; }
    sb0 = lo; hi = N_;
    while (lo < hi) { int mid = (lo + hi) >> 1; if (idx[mid] < b + 1) lo = mid + 1; else hi = mid; }
    sb1 = lo;
  }
  __syncthreads();
  int beg = sb0, end = sb1;
  const float* eo = e + o * N_;
  float m = -3.0e38f;
  for (int n = beg + t; n < end; n += 256) m = fmaxf(m, eo[n]);
  red[t] = m; __syncthreads();
  for (int s = 128; s > 0; s >>= 1) { if (t < s) red[t] = fmaxf(red[t], red[t + s]); __syncthreads(); }
  if (t == 0) mv = red[0];
  __syncthreads();
  m = mv;
  float s = 0.f;
  for (int n = beg + t; n < end; n += 256) s += __expf(eo[n] - m);
  red[t] = s; __syncthreads();
  for (int st = 128; st > 0; st >>= 1) { if (t < st) red[t] += red[t + st]; __syncthreads(); }
  if (t == 0) sv = red[0];
  __syncthreads();
  float inv = (sv > 0.f) ? 1.0f / sv : 0.f;
  float acc = 0.f;
  for (int n = beg; n < end; ++n) {
    float w = __expf(eo[n] - m) * inv;
    acc += w * feats[(size_t)n * D_ + t];
  }
  srg[((size_t)(o * B_ + b)) * D_ + t] = acc;
}

// ---------------- 8: sr = normalize(cat([fvs, srg]) @ Wsr[o]) -> bf16 ----------------
__global__ void k_sr(const float* __restrict__ fvs, const float* __restrict__ srg,
                     const float* __restrict__ Wsr, unsigned short* __restrict__ srb) {
  int b = blockIdx.x, o = blockIdx.y, t = threadIdx.x;
  __shared__ float cat[2 * D_];
  __shared__ float red[256];
  size_t rb = ((size_t)(o * B_ + b)) * D_;
  cat[t]       = fvs[rb + t];
  cat[D_ + t]  = srg[rb + t];
  __syncthreads();
  const float* w = Wsr + (size_t)o * 2 * D_ * D_;
  float acc = 0.f;
  for (int f = 0; f < 2 * D_; ++f) acc += cat[f] * w[(size_t)f * D_ + t];
  red[t] = acc * acc; __syncthreads();
  for (int s = 128; s > 0; s >>= 1) { if (t < s) red[t] += red[t + s]; __syncthreads(); }
  float scale = 1.0f / fmaxf(sqrtf(red[0]), 1e-12f);
  srb[rb + t] = f2bf(acc * scale);
}

// ---------------- 9: logits = sr @ target^T, stored as out[b][o][v] ----------------
__global__ void __launch_bounds__(256) k_logits(
    const unsigned short* __restrict__ srb, const unsigned short* __restrict__ tb,
    float* __restrict__ out) {
  int wave = threadIdx.x >> 5;
  int lane = threadIdx.x & 31;
  int wm = wave >> 2, wn = wave & 3;          // 2x4 wave grid
  int m0 = blockIdx.y * 64 + wm * 32;         // 32 rows / wave
  int n0 = blockIdx.x * 256 + wn * 64;        // 64 cols / wave

  v8f z = {0.f,0.f,0.f,0.f,0.f,0.f,0.f,0.f};
  v8f c[2][4];
  #pragma unroll
  for (int i = 0; i < 2; ++i)
    #pragma unroll
    for (int j = 0; j < 4; ++j) c[i][j] = z;

  // software-pipelined, double-buffered fragment loads
  Frag a[2][2], bfr[2][4];
  loadA(srb, D_, m0,      0, a[0][0]);
  loadA(srb, D_, m0 + 16, 0, a[0][1]);
  #pragma unroll
  for (int j = 0; j < 4; ++j) loadB(tb, D_, n0 + j * 16, 0, V_, bfr[0][j]);

  #pragma unroll
  for (int kk = 0; kk < 8; ++kk) {
    int cur = kk & 1, nxt = cur ^ 1;
    if (kk < 7) {
      int k = (kk + 1) << 5;
      loadA(srb, D_, m0,      k, a[nxt][0]);
      loadA(srb, D_, m0 + 16, k, a[nxt][1]);
      #pragma unroll
      for (int j = 0; j < 4; ++j) loadB(tb, D_, n0 + j * 16, k, V_, bfr[nxt][j]);
    }
    #pragma unroll
    for (int i = 0; i < 2; ++i)
      #pragma unroll
      for (int j = 0; j < 4; ++j) c[i][j] = WMMA_BF16(a[cur][i], bfr[cur][j], c[i][j]);
  }

  // Pin the pipeline: keep >=1 K-step of loads (12 b128) in flight ahead of
  // each WMMA pack. 96 VMEM reads total, 64 WMMAs total.
  SGB(0x020, 24);                     // loads for k0 + k1
  #pragma unroll
  for (int kk = 0; kk < 8; ++kk) {
    SGB(0x008, 8);                    // WMMAs of step kk
    if (kk < 6) SGB(0x020, 12);       // loads for step kk+2
  }

  int nn  = lane & 15;
  int hi8 = (lane >> 4) << 3;
  #pragma unroll
  for (int i = 0; i < 2; ++i)
    #pragma unroll
    for (int j = 0; j < 4; ++j)
      #pragma unroll
      for (int r = 0; r < 8; ++r) {
        int m = m0 + i * 16 + r + hi8;        // 0..1535
        int v = n0 + j * 16 + nn;
        if (v < V_) {
          int o = m >> 9;                     // m / 512
          int b = m & 511;
          // streaming output: non-temporal so the bf16 target table stays in L2
          __builtin_nontemporal_store(c[i][j][r], &out[((size_t)(b * O_ + o)) * V_ + v]);
        }
      }
}

// ---------------- launch ----------------
extern "C" void kernel_launch(void* const* d_in, const int* in_sizes, int n_in,
                              void* d_out, int out_size, void* d_ws, size_t ws_size,
                              hipStream_t stream) {
  const float* emb  = (const float*)d_in[0];
  const float* Wu   = (const float*)d_in[1];
  const float* bu   = (const float*)d_in[2];
  const float* Wv   = (const float*)d_in[3];
  const float* we   = (const float*)d_in[4];
  const float* Wsr  = (const float*)d_in[5];
  const int*   iid  = (const int*)d_in[6];
  const int*   idx  = (const int*)d_in[7];
  const int*   last = (const int*)d_in[8];
  (void)in_sizes; (void)n_in; (void)out_size; (void)ws_size;

  char* ws = (char*)d_ws;
  unsigned short* tb     = (unsigned short*)(ws + WS_TARGET);
  float*          featsf = (float*)(ws + WS_FEATSF);
  unsigned short* featsb = (unsigned short*)(ws + WS_FEATSB);
  unsigned short* wut    = (unsigned short*)(ws + WS_WUT);
  float*          fvs    = (float*)(ws + WS_FVS);
  float*          fv     = (float*)(ws + WS_FV);
  float*          e      = (float*)(ws + WS_E);
  float*          srg    = (float*)(ws + WS_SRG);
  unsigned short* srb    = (unsigned short*)(ws + WS_SRB);

  k_target<<<V_, 256, 0, stream>>>(emb, tb);
  k_feats<<<N_, 256, 0, stream>>>(emb, iid, featsf, featsb);
  k_wt<<<O_ * 256, 256, 0, stream>>>(Wu, wut);
  k_featv<<<dim3(B_, O_), 256, 0, stream>>>(featsf, Wv, last, fvs, fv);
  k_zero<<<(O_ * N_ + 255) / 256, 256, 0, stream>>>(e, O_ * N_);
  k_e<<<dim3(N_ / 16, O_), 128, 0, stream>>>(featsb, wut, bu, we, fv, idx, e);
  k_srg<<<dim3(B_, O_), 256, 0, stream>>>(e, idx, featsf, srg);
  k_sr<<<dim3(B_, O_), 256, 0, stream>>>(fvs, srg, Wsr, srb);
  k_logits<<<dim3((V_ + 255) / 256, (O_ * B_) / 64), 256, 0, stream>>>(srb, tb, (float*)d_out);
}
